// EntityEncoder_84344567759619
// MI455X (gfx1250) — matrix operand order repaired and checked by reading, not compiled, not run on hardware
//
#include <hip/hip_runtime.h>
#include <math.h>

typedef __attribute__((ext_vector_type(16))) _Float16 v16h;
typedef __attribute__((ext_vector_type(8)))  _Float16 v8h;
typedef __attribute__((ext_vector_type(8)))  float    v8f;

#define EMBED 128
#define BATCH 1024
#define MAXN  200
#define PAD   100000
#define LN_EPS 1e-5f

// ---------------------------------------------------------------------------
// WMMA fragment loaders (wave32); contiguous k order, same for A and B.
// A (16x32 f16, row-major, ld=128): lanes 0-15 hold K = k0+0..15,
// lanes 16-31 hold K = k0+16..31, row = m0 + (lane&15).
// B (32x16 f16) from a transposed copy Bt[n][k] (ld=128): lane holds column
// n0+(lane&15); lanes 0-15 K=k0+0..15, lanes 16-31 K=k0+16..31 (contig 32B).
// C/D (16x16 f32): VGPR r -> row m0 + r + (lane>=16 ? 8 : 0), col n0+(lane&15).
// ---------------------------------------------------------------------------
__device__ inline v16h load_a_frag(const _Float16* __restrict__ A, int m0, int k0, int lane) {
  const int row  = m0 + (lane & 15);
  const int koff = k0 + ((lane >> 4) << 4);
  const _Float16* p = A + row * EMBED + koff;
  v8h lo = *(const v8h*)(p);
  v8h hi = *(const v8h*)(p + 8);
  v16h a;
#pragma unroll
  for (int i = 0; i < 8; ++i) { a[i] = lo[i]; a[i + 8] = hi[i]; }
  return a;
}

__device__ inline v16h load_b_frag(const _Float16* __restrict__ Bt, int n0, int k0, int lane) {
  const int col   = n0 + (lane & 15);
  const int kbase = k0 + ((lane >> 4) << 4);
  return *(const v16h*)(Bt + col * EMBED + kbase);
}

// ---------------------------------------------------------------------------
// Kernel 1: fused weight prep (blocks 0..63) + head gathers (blocks 64..575).
// Wbil stored transposed (Bt layout); Wtail/Whead used as W^T operands, so
// their row-major storage already IS the Bt layout.
// ---------------------------------------------------------------------------
__global__ __launch_bounds__(256) void prep_heads_kernel(
    const float* __restrict__ Wbil, const float* __restrict__ Wtail,
    const float* __restrict__ Whead, const int* __restrict__ entity,
    const float* __restrict__ emb,
    _Float16* __restrict__ bilT16, _Float16* __restrict__ wt16,
    _Float16* __restrict__ wh16,
    float* __restrict__ hl, float* __restrict__ hr,
    _Float16* __restrict__ hl16, _Float16* __restrict__ hr16,
    _Float16* __restrict__ wr16) {
  const int blk = blockIdx.x;
  if (blk < 64) {
    const int i = blk * 256 + threadIdx.x;                // 0..16383
    const int e = i >> 7, d = i & 127;
    bilT16[i] = (_Float16)Wbil[d * EMBED + e];            // transpose
    wt16[i]   = (_Float16)Wtail[i];
    wh16[i]   = (_Float16)Whead[i];
  } else {
    const int b = (blk - 64) * 2 + (threadIdx.x >> 7);    // 2 rows per block
    const int t = threadIdx.x & 127;
    const int e0 = entity[b * 2 + 0], e1 = entity[b * 2 + 1];
    const float a = emb[(size_t)e0 * EMBED + t];
    const float c = emb[(size_t)e1 * EMBED + t];
    hl[b * EMBED + t] = a;  hr[b * EMBED + t] = c;
    hl16[b * EMBED + t] = (_Float16)a;
    hr16[b * EMBED + t] = (_Float16)c;
    wr16[b * EMBED + t] = (_Float16)(c - a);
  }
}

// ---------------------------------------------------------------------------
// Kernel 2: q = weak_rel @ W_bil  (1024x128x128) via WMMA.
// 64 blocks x 8 waves; wave w owns 16x16 tile (m0=16*bid, n0=16*w).
// ---------------------------------------------------------------------------
__global__ __launch_bounds__(256) void gemm_q_kernel(
    const _Float16* __restrict__ A, const _Float16* __restrict__ Bt,
    float* __restrict__ C) {
  const int lane = threadIdx.x & 31;
  const int w    = threadIdx.x >> 5;
  const int m0   = blockIdx.x * 16;
  const int n0   = w * 16;
  v8f c = {};
#pragma unroll
  for (int k0 = 0; k0 < EMBED; k0 += 32) {
    v16h a = load_a_frag(A, m0, k0, lane);
    v16h b = load_b_frag(Bt, n0, k0, lane);
    c = __builtin_amdgcn_wmma_f32_16x16x32_f16(false, a, false, b, (short)0, c, false, false);
  }
  const int row = m0 + ((lane >> 4) << 3);
  const int col = n0 + (lane & 15);
#pragma unroll
  for (int r = 0; r < 8; ++r) C[(row + r) * EMBED + col] = c[r];
}

// ---------------------------------------------------------------------------
// Kernel 3: attention gather. grid (BATCH, 2): y=0 left, y=1 right.
// One block per batch row; 8 waves x 25 neighbors, wave-per-neighbor
// 512B-coalesced float4 gathers (emb table is L2-resident: 51MB < 192MB).
// ---------------------------------------------------------------------------
__global__ __launch_bounds__(256) void attn_kernel(
    const int* __restrict__ conn_l, const int* __restrict__ conn_r,
    const float* __restrict__ emb, const float* __restrict__ q,
    _Float16* __restrict__ aggL, _Float16* __restrict__ aggR) {
  __shared__ float sc[256];
  __shared__ int   ids[256];
  __shared__ float red[256];
  __shared__ float part[8 * EMBED];

  const int side = blockIdx.y;
  const int* __restrict__ conn = side ? conn_r : conn_l;
  _Float16* __restrict__ agg16 = side ? aggR : aggL;

  const int b = blockIdx.x, t = threadIdx.x;
  const int lane = t & 31, w = t >> 5;
  const float NEG = -__builtin_inff();

  const float4 q4 = *(const float4*)(q + b * EMBED + lane * 4);

  // pass 1: scores
  for (int m = w; m < MAXN; m += 8) {
    const int rel = conn[(b * MAXN + m) * 2 + 0];
    const int ent = conn[(b * MAXN + m) * 2 + 1];
    float s;
    if (rel == PAD) {
      s = NEG;
    } else {
      const float4 r4 = *(const float4*)(emb + (size_t)rel * EMBED + lane * 4);
      float p = q4.x * r4.x + q4.y * r4.y + q4.z * r4.z + q4.w * r4.w;
      p += __shfl_xor(p, 16, 32);
      p += __shfl_xor(p, 8, 32);
      p += __shfl_xor(p, 4, 32);
      p += __shfl_xor(p, 2, 32);
      p += __shfl_xor(p, 1, 32);
      s = p;
    }
    if (lane == 0) { sc[m] = s; ids[m] = ent; }
  }
  __syncthreads();

  // softmax max
  red[t] = (t < MAXN) ? sc[t] : NEG;
  __syncthreads();
#pragma unroll
  for (int s = 128; s > 0; s >>= 1) {
    if (t < s) red[t] = fmaxf(red[t], red[t + s]);
    __syncthreads();
  }
  const float mx = red[0];
  __syncthreads();

  // exp weights + sum
  float e = 0.f;
  if (t < MAXN) {
    const float sv = sc[t];
    e = (sv == NEG) ? 0.f : __expf(sv - mx);
  }
  red[t] = e;
  __syncthreads();
#pragma unroll
  for (int s = 128; s > 0; s >>= 1) {
    if (t < s) red[t] += red[t + s];
    __syncthreads();
  }
  const float sum = red[0];
  if (t < MAXN) sc[t] = e;        // reuse sc as exp-weight array
  __syncthreads();

  // pass 2: exp-weighted tail accumulation
  float4 acc = make_float4(0.f, 0.f, 0.f, 0.f);
  for (int m = w; m < MAXN; m += 8) {
    const float wgt = sc[m];
    if (wgt != 0.f) {
      const int ent = ids[m];
      const float4 t4 = *(const float4*)(emb + (size_t)ent * EMBED + lane * 4);
      acc.x += wgt * t4.x; acc.y += wgt * t4.y;
      acc.z += wgt * t4.z; acc.w += wgt * t4.w;
    }
  }
  float* pp = &part[w * EMBED + lane * 4];
  pp[0] = acc.x; pp[1] = acc.y; pp[2] = acc.z; pp[3] = acc.w;
  __syncthreads();

  if (t < EMBED) {
    float s = 0.f;
#pragma unroll
    for (int ww = 0; ww < 8; ++ww) s += part[ww * EMBED + t];
    agg16[b * EMBED + t] = (_Float16)(s / sum);
  }
}

// ---------------------------------------------------------------------------
// Kernel 4: fused branch + LayerNorm. grid (64, 2).
// x = relu(agg @ Wtail^T + head @ Whead^T) + head, staged in LDS (16x128),
// then per-row LN (16 threads/row, half-wave shfl reduce) -> d_out.
// ---------------------------------------------------------------------------
__global__ __launch_bounds__(256) void branch_ln_kernel(
    const _Float16* __restrict__ aggL, const _Float16* __restrict__ aggR,
    const _Float16* __restrict__ hl16, const _Float16* __restrict__ hr16,
    const _Float16* __restrict__ Wt, const _Float16* __restrict__ Wh,
    const float* __restrict__ hlF, const float* __restrict__ hrF,
    const float* __restrict__ gamma, const float* __restrict__ beta,
    float* __restrict__ out) {
  __shared__ float xs[16 * EMBED];   // 8 KB

  const int side = blockIdx.y;
  const _Float16* __restrict__ Agg    = side ? aggR : aggL;
  const _Float16* __restrict__ Head16 = side ? hr16 : hl16;
  const float*    __restrict__ HeadF  = side ? hrF  : hlF;
  float* __restrict__ o = out + (size_t)side * BATCH * EMBED;

  const int lane = threadIdx.x & 31;
  const int w    = threadIdx.x >> 5;
  const int m0   = blockIdx.x * 16;
  const int n0   = w * 16;

  v8f c = {};
#pragma unroll
  for (int k0 = 0; k0 < EMBED; k0 += 32) {
    v16h a  = load_a_frag(Agg, m0, k0, lane);
    v16h b  = load_b_frag(Wt, n0, k0, lane);
    c = __builtin_amdgcn_wmma_f32_16x16x32_f16(false, a, false, b, (short)0, c, false, false);
    v16h a2 = load_a_frag(Head16, m0, k0, lane);
    v16h b2 = load_b_frag(Wh, n0, k0, lane);
    c = __builtin_amdgcn_wmma_f32_16x16x32_f16(false, a2, false, b2, (short)0, c, false, false);
  }
  const int rbase = (lane >> 4) << 3;                 // 0 or 8 (tile-local)
  const int col   = n0 + (lane & 15);
#pragma unroll
  for (int r = 0; r < 8; ++r) {
    const float hd = HeadF[(m0 + rbase + r) * EMBED + col];
    xs[(rbase + r) * EMBED + col] = fmaxf(c[r], 0.f) + hd;
  }
  __syncthreads();

  // LayerNorm: thread t handles row (t>>4), columns (t&15)*8 .. +7.
  const int row = threadIdx.x >> 4;                   // 0..15
  const int sub = threadIdx.x & 15;                   // 0..15
  const float* xr = &xs[row * EMBED + sub * 8];

  float s = 0.f;
#pragma unroll
  for (int i = 0; i < 8; ++i) s += xr[i];
  s += __shfl_xor(s, 8, 32);
  s += __shfl_xor(s, 4, 32);
  s += __shfl_xor(s, 2, 32);
  s += __shfl_xor(s, 1, 32);
  const float mu = s * (1.f / EMBED);

  float v = 0.f;
#pragma unroll
  for (int i = 0; i < 8; ++i) { const float d = xr[i] - mu; v += d * d; }
  v += __shfl_xor(v, 8, 32);
  v += __shfl_xor(v, 4, 32);
  v += __shfl_xor(v, 2, 32);
  v += __shfl_xor(v, 1, 32);
  const float inv = rsqrtf(v * (1.f / EMBED) + LN_EPS);

#pragma unroll
  for (int i = 0; i < 8; ++i) {
    const int cidx = sub * 8 + i;
    o[(m0 + row) * EMBED + cidx] = (xr[i] - mu) * inv * gamma[cidx] + beta[cidx];
  }
}

// ---------------------------------------------------------------------------
extern "C" void kernel_launch(void* const* d_in, const int* in_sizes, int n_in,
                              void* d_out, int out_size, void* d_ws, size_t ws_size,
                              hipStream_t stream) {
  const int*   entity     = (const int*)d_in[0];
  const int*   conn_left  = (const int*)d_in[1];
  const int*   conn_right = (const int*)d_in[2];
  const float* emb        = (const float*)d_in[3];
  const float* W_bil      = (const float*)d_in[4];
  const float* W_tail     = (const float*)d_in[5];
  const float* W_head     = (const float*)d_in[6];
  const float* gamma      = (const float*)d_in[7];
  const float* beta       = (const float*)d_in[8];
  float* out = (float*)d_out;

  // workspace layout (bytes), 256B aligned, ~2.98 MB total
  char* ws = (char*)d_ws;
  _Float16* bilT16 = (_Float16*)(ws + 0);          //  32 KB
  _Float16* wt16   = (_Float16*)(ws + 32768);      //  32 KB
  _Float16* wh16   = (_Float16*)(ws + 65536);      //  32 KB
  float*    hl     = (float*)   (ws + 98304);      // 512 KB
  float*    hr     = (float*)   (ws + 622592);     // 512 KB
  _Float16* hl16   = (_Float16*)(ws + 1146880);    // 256 KB
  _Float16* hr16   = (_Float16*)(ws + 1409024);    // 256 KB
  _Float16* wr16   = (_Float16*)(ws + 1671168);    // 256 KB
  float*    qbuf   = (float*)   (ws + 1933312);    // 512 KB
  _Float16* aggL16 = (_Float16*)(ws + 2457600);    // 256 KB
  _Float16* aggR16 = (_Float16*)(ws + 2719744);    // 256 KB

  prep_heads_kernel<<<576, 256, 0, stream>>>(
      W_bil, W_tail, W_head, entity, emb,
      bilT16, wt16, wh16, hl, hr, hl16, hr16, wr16);
  gemm_q_kernel<<<64, 256, 0, stream>>>(wr16, bilT16, qbuf);
  attn_kernel<<<dim3(BATCH, 2), 256, 0, stream>>>(
      conn_left, conn_right, emb, qbuf, aggL16, aggR16);
  branch_ln_kernel<<<dim3(64, 2), 256, 0, stream>>>(
      aggL16, aggR16, hl16, hr16, wt16, wh16, hl, hr, gamma, beta, out);
}
